// CostVolume_45938970198692
// MI455X (gfx1250) — compile-verified, matrix operand order checked by reading, not run on hardware
//
#include <hip/hip_runtime.h>
#include <hip/hip_bf16.h>
#include <math.h>

typedef __attribute__((ext_vector_type(16))) _Float16 v16h;
typedef __attribute__((ext_vector_type(8)))  _Float16 v8h;
typedef __attribute__((ext_vector_type(8)))  float    v8f;

#define B_ 4
#define S_ 4096
#define N_ 4096
#define C_ 64
#define KQ 32
#define KS 16

#define PTS1 4
#define COLS1 (PTS1 * KQ)   // 128 columns per workgroup (stage 1)
#define KP1 160             // 138 -> 160 (multiple of 32)

#define PTS2 8
#define COLS2 (PTS2 * KS)   // 128 columns per workgroup (stage 2)
#define KP2 192             // mlp3 input is exactly 192

// ---------------------------------------------------------------------------
// Brute-force KNN: one query per thread, refs staged through LDS in tiles of
// 256. Rank by |r|^2 - 2 q.r (|q|^2 constant per query). Sorted top-K kept by
// insertion in scratch; earliest index wins ties (strict <).
// ---------------------------------------------------------------------------
template<int K, int NR>
__global__ void knn_kernel(const float* __restrict__ q,
                           const float* __restrict__ r,
                           int* __restrict__ outIdx) {
  __shared__ float sx[256], sy[256], sz[256];
  const int tid = threadIdx.x;
  const int gq  = blockIdx.x * 256 + tid;   // grid = B_*S_/256 -> one b per block
  const int b   = gq / S_;
  const int s   = gq % S_;
  const float qx = q[(b * 3 + 0) * S_ + s];
  const float qy = q[(b * 3 + 1) * S_ + s];
  const float qz = q[(b * 3 + 2) * S_ + s];
  float bd[K]; int bi[K];
#pragma unroll
  for (int i = 0; i < K; ++i) { bd[i] = 3.4e38f; bi[i] = 0; }
  for (int t0 = 0; t0 < NR; t0 += 256) {
    __syncthreads();
    sx[tid] = r[(b * 3 + 0) * NR + t0 + tid];
    sy[tid] = r[(b * 3 + 1) * NR + t0 + tid];
    sz[tid] = r[(b * 3 + 2) * NR + t0 + tid];
    __syncthreads();
    for (int j = 0; j < 256; ++j) {
      const float rx = sx[j], ry = sy[j], rz = sz[j];
      const float d = rx * rx + ry * ry + rz * rz
                    - 2.f * (qx * rx + qy * ry + qz * rz);
      if (d < bd[K - 1]) {
        int p = K - 1;
        while (p > 0 && d < bd[p - 1]) { bd[p] = bd[p - 1]; bi[p] = bi[p - 1]; --p; }
        bd[p] = d; bi[p] = t0 + j;
      }
    }
  }
#pragma unroll
  for (int i = 0; i < K; ++i) outIdx[(size_t)(b * S_ + s) * K + i] = bi[i];
}

// ---------------------------------------------------------------------------
// Stage a layer's weights into LDS as zero-padded f16 [COUT][KP] + bias f32.
// Done once per block per layer; kills per-fragment cvt/guard VALU work.
// ---------------------------------------------------------------------------
template<int CIN, int KP, int COUT>
__device__ __forceinline__ void stage_weights(
    const float* __restrict__ W, const float* __restrict__ bias,
    _Float16* __restrict__ WB, float* __restrict__ Bb, int tid)
{
  __builtin_prefetch(W, 0, 1);   // global_prefetch_b8 on the weight stream
  for (int i = tid; i < COUT * KP; i += 256) {
    const int rr = i / KP;
    const int k  = i - rr * KP;
    WB[i] = (k < CIN) ? (_Float16)W[rr * CIN + k] : (_Float16)0.f;
  }
  if (tid < COUT) Bb[tid] = bias[tid];
}

// ---------------------------------------------------------------------------
// Y[:, yoff..yoff+COUT) = relu(W @ X + b) via v_wmma_f32_16x16x32_f16.
// Layouts are fragment-native (everything contiguous 32B/16B LDS accesses):
//   WB : f16 [COUT][KP]   (A frag: lane M=m0+l15 reads K-chunk -> 1x v16h)
//   Xs : f16 [COLS][LDX]  (B frag: lane N=col reads K-chunk    -> 1x v16h)
//   Ys : f16 [COLS][LDY]  (D frag: lane stores 8 halves        -> 1x v8h)
// 8 waves == 8 column tiles; K- and M-loops fully unrolled (compile-time).
// ---------------------------------------------------------------------------
template<int COLS, int KP, int COUT, int LDX, int LDY>
__device__ __forceinline__ void gemm_lds(
    const _Float16* __restrict__ WB, const float* __restrict__ Bb,
    const _Float16* __restrict__ Xs, _Float16* __restrict__ Ys,
    int yoff, int lane, int wave)
{
  const int half = lane >> 4;       // K offset 0/16 within a 32-chunk
  const int l15  = lane & 15;
  const int bcol = (wave << 4) + l15;
  const v16h* bp = (const v16h*)(Xs + (size_t)bcol * LDX);
#pragma unroll
  for (int m0 = 0; m0 < COUT; m0 += 16) {
    const v16h* ap = (const v16h*)(WB + (size_t)(m0 + l15) * KP);
    v8f acc = {};
#pragma unroll
    for (int k0 = 0; k0 < KP; k0 += 32) {
      const int kidx = (k0 >> 4) + half;
      acc = __builtin_amdgcn_wmma_f32_16x16x32_f16(
          false, ap[kidx], false, bp[kidx], (short)0, acc, false, false);
    }
    v8h o;
#pragma unroll
    for (int rr = 0; rr < 8; ++rr) {
      const float y = acc[rr] + Bb[m0 + rr + 8 * half];
      o[rr] = (_Float16)(y > 0.f ? y : 0.f);
    }
    *(v8h*)(Ys + (size_t)bcol * LDY + yoff + m0 + 8 * half) = o;
  }
}

// ---------------------------------------------------------------------------
// Stage 1: gather 32 NN from cloud 2; mlp1 (138->128->64->64), xyz1 (10->64),
// mlp2 (128->128->64); softmax over k=32; weighted pool -> pooled (B,64,S).
// Tile: 4 points x 32 neighbors = 128 cols, 256 threads (8 waves).
// LDS (f16 halves): A 128x160 | Bf 128x128 | Cf 128x64 | XC 128x32 |
//                   Pc 128x64 | WB 128x160 | bias f32[128]  ~= 153 KB
// ---------------------------------------------------------------------------
__global__ void stage1_kernel(
    const float* __restrict__ xyz1, const float* __restrict__ f1,
    const float* __restrict__ xyz2, const float* __restrict__ f2,
    const int* __restrict__ idxq,
    const float* __restrict__ W1, const float* __restrict__ b1,
    const float* __restrict__ W2, const float* __restrict__ b2,
    const float* __restrict__ W3, const float* __restrict__ b3,
    const float* __restrict__ Wx, const float* __restrict__ bx,
    const float* __restrict__ Wa, const float* __restrict__ ba,
    const float* __restrict__ Wb, const float* __restrict__ bb,
    float* __restrict__ pooled)
{
  extern __shared__ char smem[];
  _Float16* A  = (_Float16*)smem;            // [128][160]  X0 / mlp2 hidden
  _Float16* Bf = A  + COLS1 * KP1;           // [128][128]  Y1 / mlp2 input
  _Float16* Cf = Bf + COLS1 * 128;           // [128][64]   Y2
  _Float16* XC = Cf + COLS1 * 64;            // [128][32]   xyz cat (padded)
  _Float16* Pc = XC + COLS1 * 32;            // [128][64]   pi_concat
  _Float16* WB = Pc + COLS1 * 64;            // [128][160]  staged weights
  float*    Bb = (float*)(WB + COLS1 * KP1); // f32[128]    staged bias
  const int tid  = threadIdx.x;
  const int lane = tid & 31, wave = tid >> 5;
  const int b  = blockIdx.x / (S_ / PTS1);
  const int s0 = (blockIdx.x % (S_ / PTS1)) * PTS1;

  // Build X0 (col-major [col][160], rows 138..159 zeroed) + xyz-cat copy.
  for (int i = tid; i < COLS1 * KP1; i += 256) {
    const int col = i / KP1;
    const int row = i - col * KP1;
    const int p  = col >> 5;
    const int kk = col & (KQ - 1);
    const int s  = s0 + p;
    float v = 0.f;
    if (row < 138) {
      const int nb = idxq[(size_t)(b * S_ + s) * KQ + kk];
      if (row < 10) {
        const float px = xyz1[(b * 3 + 0) * S_ + s];
        const float py = xyz1[(b * 3 + 1) * S_ + s];
        const float pz = xyz1[(b * 3 + 2) * S_ + s];
        const float nx = xyz2[(b * 3 + 0) * N_ + nb];
        const float ny = xyz2[(b * 3 + 1) * N_ + nb];
        const float nz = xyz2[(b * 3 + 2) * N_ + nb];
        const float dx = nx - px, dy = ny - py, dz = nz - pz;
        switch (row) {
          case 0: v = px; break; case 1: v = py; break; case 2: v = pz; break;
          case 3: v = nx; break; case 4: v = ny; break; case 5: v = nz; break;
          case 6: v = dx; break; case 7: v = dy; break; case 8: v = dz; break;
          default: v = sqrtf(dx * dx + dy * dy + dz * dz + 1e-20f); break;
        }
      } else if (row < 74) {
        v = f1[((size_t)b * C_ + (row - 10)) * S_ + s];
      } else {
        v = f2[((size_t)b * C_ + (row - 74)) * N_ + nb];
      }
    }
    A[i] = (_Float16)v;
    if (row < 32) XC[col * 32 + row] = (row < 10) ? (_Float16)v : (_Float16)0.f;
  }

  // mlp1 L1: 138(->160) -> 128
  __syncthreads(); stage_weights<138, KP1, 128>(W1, b1, WB, Bb, tid);
  __syncthreads(); gemm_lds<COLS1, KP1, 128, KP1, 128>(WB, Bb, A, Bf, 0, lane, wave);
  // mlp1 L2: 128 -> 64
  __syncthreads(); stage_weights<128, 128, 64>(W2, b2, WB, Bb, tid);
  __syncthreads(); gemm_lds<COLS1, 128, 64, 128, 64>(WB, Bb, Bf, Cf, 0, lane, wave);
  // mlp1 L3: 64 -> 64 (feat1_new -> Bf channels 64..127)
  __syncthreads(); stage_weights<64, 64, 64>(W3, b3, WB, Bb, tid);
  __syncthreads(); gemm_lds<COLS1, 64, 64, 64, 128>(WB, Bb, Cf, Bf, 64, lane, wave);
  // xyz1 enc: 10(->32) -> 64 (-> Bf channels 0..63); Bf is now mlp2 input
  __syncthreads(); stage_weights<10, 32, 64>(Wx, bx, WB, Bb, tid);
  __syncthreads(); gemm_lds<COLS1, 32, 64, 32, 128>(WB, Bb, XC, Bf, 0, lane, wave);
  // mlp2 L1: 128 -> 128 (-> A, stride 160)
  __syncthreads(); stage_weights<128, 128, 128>(Wa, ba, WB, Bb, tid);
  __syncthreads(); gemm_lds<COLS1, 128, 128, 128, KP1>(WB, Bb, Bf, A, 0, lane, wave);
  // mlp2 L2: 128 -> 64 (pi_concat -> Pc)
  __syncthreads(); stage_weights<128, 128, 64>(Wb, bb, WB, Bb, tid);
  __syncthreads(); gemm_lds<COLS1, 128, 64, KP1, 64>(WB, Bb, A, Pc, 0, lane, wave);
  __syncthreads();

  // Per-(channel, point) softmax over 32 neighbors + weighted pool.
  {
    const int c = tid & 63;
    const int p = tid >> 6;            // 0..3
    float mx = -3.4e38f;
    for (int j = 0; j < KQ; ++j) {
      const float x = (float)Pc[(p * KQ + j) * 64 + c];
      mx = x > mx ? x : mx;
    }
    float sum = 0.f, acc = 0.f;
    for (int j = 0; j < KQ; ++j) {
      const float e = __expf((float)Pc[(p * KQ + j) * 64 + c] - mx);
      sum += e;
      acc += e * (float)Bf[(p * KQ + j) * 128 + 64 + c];
    }
    pooled[((size_t)b * C_ + c) * S_ + (s0 + p)] = acc / sum;
  }
}

// ---------------------------------------------------------------------------
// Stage 2: self-KNN(16) gather; xyz2 (10->64), mlp3 (192->128->64);
// softmax over k=16; weighted pool -> out (B,64,S).
// Tile: 8 points x 16 neighbors = 128 cols.
// LDS: A2 128x192 | XC2 128x32 | Y2 128x128 | P2 128x64 | WB2 128x192 ~= 153KB
// ---------------------------------------------------------------------------
__global__ void stage2_kernel(
    const float* __restrict__ xyz1, const float* __restrict__ f1,
    const float* __restrict__ pooled, const int* __restrict__ idxs,
    const float* __restrict__ Wx, const float* __restrict__ bx,
    const float* __restrict__ Wa, const float* __restrict__ ba,
    const float* __restrict__ Wb, const float* __restrict__ bb,
    float* __restrict__ out)
{
  extern __shared__ char smem[];
  _Float16* A2 = (_Float16*)smem;             // [128][192]  mlp3 input
  _Float16* XC = A2 + COLS2 * KP2;            // [128][32]
  _Float16* Y2 = XC + COLS2 * 32;             // [128][128]
  _Float16* P2 = Y2 + COLS2 * 128;            // [128][64]   pc_concat
  _Float16* WB = P2 + COLS2 * 64;             // [128][192]
  float*    Bb = (float*)(WB + COLS2 * KP2);  // f32[128]
  const int tid  = threadIdx.x;
  const int lane = tid & 31, wave = tid >> 5;
  const int b  = blockIdx.x / (S_ / PTS2);
  const int s0 = (blockIdx.x % (S_ / PTS2)) * PTS2;

  // xyz cat (channels 0..9, padded to 32), col-major
  for (int i = tid; i < COLS2 * 32; i += 256) {
    const int col = i >> 5;
    const int row = i & 31;
    const int p  = col >> 4;
    const int kk = col & (KS - 1);
    const int s  = s0 + p;
    float v = 0.f;
    if (row < 10) {
      const int nb = idxs[(size_t)(b * S_ + s) * KS + kk];
      const float px = xyz1[(b * 3 + 0) * S_ + s];
      const float py = xyz1[(b * 3 + 1) * S_ + s];
      const float pz = xyz1[(b * 3 + 2) * S_ + s];
      const float nx = xyz1[(b * 3 + 0) * S_ + nb];
      const float ny = xyz1[(b * 3 + 1) * S_ + nb];
      const float nz = xyz1[(b * 3 + 2) * S_ + nb];
      const float dx = nx - px, dy = ny - py, dz = nz - pz;
      switch (row) {
        case 0: v = px; break; case 1: v = py; break; case 2: v = pz; break;
        case 3: v = nx; break; case 4: v = ny; break; case 5: v = nz; break;
        case 6: v = dx; break; case 7: v = dy; break; case 8: v = dz; break;
        default: v = sqrtf(dx * dx + dy * dy + dz * dz + 1e-20f); break;
      }
    }
    XC[i] = (_Float16)v;
  }
  // mlp3 input channels 64..127 = feature1 bc, 128..191 = grouped pooled
  for (int i = tid; i < COLS2 * 128; i += 256) {
    const int col = i >> 7;
    const int rr  = i & 127;
    const int p  = col >> 4;
    const int kk = col & (KS - 1);
    const int s  = s0 + p;
    float v;
    if (rr < 64) {
      v = f1[((size_t)b * C_ + rr) * S_ + s];
    } else {
      const int nb = idxs[(size_t)(b * S_ + s) * KS + kk];
      v = pooled[((size_t)b * C_ + (rr - 64)) * S_ + nb];
    }
    A2[(size_t)col * KP2 + 64 + rr] = (_Float16)v;
  }

  // xyz2 enc: 10(->32) -> 64 (-> A2 channels 0..63)
  __syncthreads(); stage_weights<10, 32, 64>(Wx, bx, WB, Bb, tid);
  __syncthreads(); gemm_lds<COLS2, 32, 64, 32, KP2>(WB, Bb, XC, A2, 0, lane, wave);
  // mlp3 L1: 192 -> 128
  __syncthreads(); stage_weights<192, KP2, 128>(Wa, ba, WB, Bb, tid);
  __syncthreads(); gemm_lds<COLS2, KP2, 128, KP2, 128>(WB, Bb, A2, Y2, 0, lane, wave);
  // mlp3 L2: 128 -> 64 (pc_concat)
  __syncthreads(); stage_weights<128, 128, 64>(Wb, bb, WB, Bb, tid);
  __syncthreads(); gemm_lds<COLS2, 128, 64, 128, 64>(WB, Bb, Y2, P2, 0, lane, wave);
  __syncthreads();

  for (int t = tid; t < 64 * PTS2; t += 256) {
    const int c = t & 63;
    const int p = t >> 6;              // 0..7
    float mx = -3.4e38f;
    for (int j = 0; j < KS; ++j) {
      const float x = (float)P2[(p * KS + j) * 64 + c];
      mx = x > mx ? x : mx;
    }
    float sum = 0.f, acc = 0.f;
    for (int j = 0; j < KS; ++j) {
      const float e = __expf((float)P2[(p * KS + j) * 64 + c] - mx);
      sum += e;
      acc += e * (float)A2[(size_t)(p * KS + j) * KP2 + 128 + c];
    }
    out[((size_t)b * C_ + c) * S_ + (s0 + p)] = acc / sum;
  }
}

// ---------------------------------------------------------------------------
extern "C" void kernel_launch(void* const* d_in, const int* in_sizes, int n_in,
                              void* d_out, int out_size, void* d_ws, size_t ws_size,
                              hipStream_t stream) {
  const float* xyz1 = (const float*)d_in[0];
  const float* f1   = (const float*)d_in[1];
  const float* xyz2 = (const float*)d_in[2];
  const float* f2   = (const float*)d_in[3];
  const float* W1  = (const float*)d_in[4];  const float* b1  = (const float*)d_in[5];
  const float* W2  = (const float*)d_in[6];  const float* b2  = (const float*)d_in[7];
  const float* W3  = (const float*)d_in[8];  const float* b3  = (const float*)d_in[9];
  const float* Wx1 = (const float*)d_in[10]; const float* bx1 = (const float*)d_in[11];
  const float* Wx2 = (const float*)d_in[12]; const float* bx2 = (const float*)d_in[13];
  const float* Wa  = (const float*)d_in[14]; const float* ba  = (const float*)d_in[15];
  const float* Wb  = (const float*)d_in[16]; const float* bb  = (const float*)d_in[17];
  const float* Wc  = (const float*)d_in[18]; const float* bc  = (const float*)d_in[19];
  const float* Wd  = (const float*)d_in[20]; const float* bd  = (const float*)d_in[21];

  // Workspace: idx_q (B*S*32 i32) | idx_self (B*S*16 i32) | pooled (B*64*S f32)
  int*   idxq   = (int*)d_ws;
  int*   idxs   = idxq + (size_t)B_ * S_ * KQ;
  float* pooled = (float*)(idxs + (size_t)B_ * S_ * KS);

  const dim3 blk(256);
  knn_kernel<KQ, N_><<<(B_ * S_) / 256, blk, 0, stream>>>(xyz1, xyz2, idxq);
  knn_kernel<KS, S_><<<(B_ * S_) / 256, blk, 0, stream>>>(xyz1, xyz1, idxs);

  const int lds1 =
      (COLS1 * KP1 + COLS1 * 128 + COLS1 * 64 + COLS1 * 32 + COLS1 * 64 +
       COLS1 * KP1) * (int)sizeof(_Float16) + 128 * (int)sizeof(float);
  const int lds2 =
      (COLS2 * KP2 + COLS2 * 32 + COLS2 * 128 + COLS2 * 64 +
       COLS2 * KP2) * (int)sizeof(_Float16) + 128 * (int)sizeof(float);
  (void)hipFuncSetAttribute((const void*)stage1_kernel,
                            hipFuncAttributeMaxDynamicSharedMemorySize, lds1);
  (void)hipFuncSetAttribute((const void*)stage2_kernel,
                            hipFuncAttributeMaxDynamicSharedMemorySize, lds2);

  stage1_kernel<<<B_ * (S_ / PTS1), blk, lds1, stream>>>(
      xyz1, f1, xyz2, f2, idxq,
      W1, b1, W2, b2, W3, b3, Wx1, bx1, Wa, ba, Wb, bb, pooled);
  stage2_kernel<<<B_ * (S_ / PTS2), blk, lds2, stream>>>(
      xyz1, f1, pooled, idxs,
      Wx2, bx2, Wc, bc, Wd, bd, (float*)d_out);
}
